// AttnBlock_22969485099213
// MI455X (gfx1250) — matrix operand order repaired and checked by reading, not compiled, not run on hardware
//
#include <hip/hip_runtime.h>
#include <hip/hip_bf16.h>
#include <stdint.h>

// ---------------------------------------------------------------------------
// CDNA5 (gfx1250) attention block: GN -> QKV GEMM -> flash attn -> proj+res
// All matmuls via v_wmma_f32_16x16x32_bf16, fp32 accumulate.
// GEMM tiles staged with GLOBAL_LOAD_ASYNC_TO_LDS_B128 (ASYNCcnt-tracked DMA,
// no VGPR round trip) into double-buffered LDS; WMMAs overlap async fills.
// ---------------------------------------------------------------------------

typedef __attribute__((ext_vector_type(16))) __bf16 v16bf;
typedef __attribute__((ext_vector_type(8)))  float  v8f;

union Frag {
    v16bf v;
    uint4 q[2];
    uint32_t u[8];
};

static __device__ __forceinline__ unsigned short f2bf(float f) {
    uint32_t u = __float_as_uint(f);
    uint32_t r = u + 0x7FFFu + ((u >> 16) & 1u);   // round-to-nearest-even
    return (unsigned short)(r >> 16);
}

static __device__ __forceinline__ v8f wmma_bf16(const Frag& a, const Frag& b, v8f c) {
    return __builtin_amdgcn_wmma_f32_16x16x32_bf16(
        /*neg_a=*/false, a.v, /*neg_b=*/false, b.v,
        /*c_mod=*/(short)0, c, /*reuse_a=*/false, /*reuse_b=*/false);
}

// CDNA5 async DMA: global -> LDS, 16 bytes, tracked by ASYNCcnt.
// VDST = 32-bit LDS byte address (low 32 bits of generic shared pointer,
// per ISA aperture truncation), VADDR = 64-bit global address.
static __device__ __forceinline__ void async_b128(uint32_t lds_addr, const void* gaddr) {
    asm volatile("global_load_async_to_lds_b128 %0, %1, off"
                 :: "v"(lds_addr), "v"(gaddr) : "memory");
}
static __device__ __forceinline__ void wait_async0() {
    asm volatile("s_wait_asynccnt 0x0" ::: "memory");
}
static __device__ __forceinline__ uint32_t lds_addr_of(const void* p) {
    return (uint32_t)(size_t)p;
}

// ---------------------------------------------------------------------------
// Kernel 1: fp32 -> bf16 weight conversion (grid-stride)
// ---------------------------------------------------------------------------
__global__ void __launch_bounds__(256)
cvt_bf16_kernel(const float* __restrict__ src, unsigned short* __restrict__ dst, int n) {
    for (int i = blockIdx.x * blockDim.x + threadIdx.x; i < n; i += gridDim.x * blockDim.x)
        dst[i] = f2bf(src[i]);
}

// ---------------------------------------------------------------------------
// Kernel 2: GroupNorm. One block per (batch, group). 32 groups, 16 ch/group,
// 4096 pixels. Writes Ht[b][n][c] bf16 (row-major in c, feeds WMMA A frags).
// ---------------------------------------------------------------------------
__global__ void __launch_bounds__(256)
groupnorm_kernel(const float* __restrict__ X,
                 const float* __restrict__ gw, const float* __restrict__ gb,
                 unsigned short* __restrict__ Ht) {
    __shared__ float s1[256], s2[256];
    __shared__ float mean_s, rstd_s;
    const int b = blockIdx.x >> 5;
    const int g = blockIdx.x & 31;
    const int tid = threadIdx.x;
    const size_t xbase = ((size_t)b * 512 + g * 16) * 4096;

    float sum = 0.f, sq = 0.f;
    for (int idx = tid; idx < 65536; idx += 256) {
        const int ch = idx >> 12, pix = idx & 4095;
        const float v = X[xbase + ((size_t)ch << 12) + pix];
        sum += v; sq += v * v;
    }
    s1[tid] = sum; s2[tid] = sq;
    __syncthreads();
    for (int off = 128; off; off >>= 1) {
        if (tid < off) { s1[tid] += s1[tid + off]; s2[tid] += s2[tid + off]; }
        __syncthreads();
    }
    if (tid == 0) {
        const float mean = s1[0] * (1.f / 65536.f);
        const float var  = s2[0] * (1.f / 65536.f) - mean * mean;
        mean_s = mean;
        rstd_s = rsqrtf(var + 1e-5f);
    }
    __syncthreads();
    const float mean = mean_s, rstd = rstd_s;

    for (int idx = tid; idx < 65536; idx += 256) {
        const int ch = idx >> 12, pix = idx & 4095;
        const int c = g * 16 + ch;
        const float v = X[xbase + ((size_t)ch << 12) + pix];
        const float hn = (v - mean) * rstd * gw[c] + gb[c];
        Ht[((size_t)(b * 4096 + pix)) * 512 + c] = f2bf(hn);
    }
}

// ---------------------------------------------------------------------------
// Kernel 3: WMMA GEMM  Y = A @ W^T + bias.  A: bf16 [b,4096,512], W: bf16
// [512,512] row-major.  Block = 8 waves, tile 64(M) x 128(N); wave tile
// 16 x 64.  K-steps of 32 staged into double-buffered LDS (rows padded to
// 40 elements to spread banks) by GLOBAL_LOAD_ASYNC_TO_LDS_B128: async fills
// of buffer[nxt] overlap WMMAs on buffer[cur]; s_wait_asynccnt 0 + barrier
// per k-step.
// mode 0: Y bf16 [b,n,c];  mode 2: Y bf16 [b,c,n] (transposed, for V);
// mode 1: fp32 [b,c,n] = acc + bias + resid  -> d_out (proj + residual).
// ---------------------------------------------------------------------------
#define LDA 40   // padded row length (elements) for 32-element K tiles

__global__ void __launch_bounds__(256)
gemm_kernel(const unsigned short* __restrict__ A,
            const unsigned short* __restrict__ Wb,
            const float* __restrict__ bias,
            const float* __restrict__ resid,
            unsigned short* __restrict__ Ybf,
            float* __restrict__ Yf,
            int mode) {
    __shared__ unsigned short As[2][64 * LDA];    // 2 x 5 KB
    __shared__ unsigned short Bs[2][128 * LDA];   // 2 x 10 KB

    const int b = blockIdx.z;
    const int Mbase = blockIdx.x * 64;
    const int Nbase = blockIdx.y * 128;
    const int tid = threadIdx.x, lane = tid & 31, wv = tid >> 5;
    const int waveM = wv >> 1, waveN = wv & 1;
    const int hi = lane >> 4;            // lane half
    const int mb = hi * 8;               // C/D row offset for this half
    const int nl = lane & 15;

    // cooperative tile-load mapping (16B per async op)
    const int arow_ld = tid >> 2, apart = tid & 3;   // A: 64 rows x 4x8 elems
    const int bcol_ld = tid >> 1, bpart = tid & 1;   // B: 128 cols x 2x16 elems
    const unsigned short* AgK = A + ((size_t)(b * 4096 + Mbase + arow_ld)) * 512
                                  + apart * 8;
    const unsigned short* BgK = Wb + (size_t)(Nbase + bcol_ld) * 512 + bpart * 16;

    // per-buffer LDS destinations for this thread's async ops
    uint32_t ldsA[2], ldsB0[2], ldsB1[2];
    ldsA[0]  = lds_addr_of(&As[0][arow_ld * LDA + apart * 8]);
    ldsA[1]  = lds_addr_of(&As[1][arow_ld * LDA + apart * 8]);
    ldsB0[0] = lds_addr_of(&Bs[0][bcol_ld * LDA + bpart * 16]);
    ldsB0[1] = lds_addr_of(&Bs[1][bcol_ld * LDA + bpart * 16]);
    ldsB1[0] = lds_addr_of(&Bs[0][bcol_ld * LDA + bpart * 16 + 8]);
    ldsB1[1] = lds_addr_of(&Bs[1][bcol_ld * LDA + bpart * 16 + 8]);

    v8f acc[4];
#pragma unroll
    for (int j = 0; j < 4; j++)
#pragma unroll
        for (int i = 0; i < 8; i++) acc[j][i] = 0.f;

    // prologue: async-stage k-step 0 into buffer 0
    async_b128(ldsA[0],  AgK);
    async_b128(ldsB0[0], BgK);
    async_b128(ldsB1[0], BgK + 8);
    wait_async0();
    __syncthreads();

#pragma unroll
    for (int ks = 0; ks < 16; ks++) {
        const int cur = ks & 1, nxt = cur ^ 1;
        // launch async DMA of next tile into the other buffer (no VGPR use)
        if (ks + 1 < 16) {
            const int kb = (ks + 1) * 32;
            async_b128(ldsA[nxt],  AgK + kb);
            async_b128(ldsB0[nxt], BgK + kb);
            async_b128(ldsB1[nxt], BgK + kb + 8);
            __builtin_prefetch(AgK + kb + 32, 0, 3);   // warm L2 for ks+2
        }
        // compute from LDS[cur] while DMA fills LDS[nxt]
        Frag a;
        const uint4* ap = (const uint4*)&As[cur][(waveM * 16 + nl) * LDA + (hi << 4)];
        a.q[0] = ap[0];                  // K +0..7  (or +16..23)
        a.q[1] = ap[1];                  // K +8..15 (or +24..31)
#pragma unroll
        for (int j = 0; j < 4; j++) {
            Frag bb;
            const uint4* bp =
                (const uint4*)&Bs[cur][(waveN * 64 + j * 16 + nl) * LDA + (hi << 4)];
            bb.q[0] = bp[0];
            bb.q[1] = bp[1];
            acc[j] = wmma_bf16(a, bb, acc[j]);
        }
        if (ks + 1 < 16) {
            wait_async0();       // drain this wave's DMA
            __syncthreads();     // all waves' tiles landed; swap buffers
        }
    }

#pragma unroll
    for (int j = 0; j < 4; j++) {
        const int col = Nbase + waveN * 64 + j * 16 + nl;
        const float bs = bias[col];
#pragma unroll
        for (int v = 0; v < 8; v++) {
            const int row = Mbase + waveM * 16 + v + mb;
            const float val = acc[j][v] + bs;
            if (mode == 0) {
                Ybf[((size_t)(b * 4096 + row)) * 512 + col] = f2bf(val);
            } else if (mode == 2) {
                Ybf[((size_t)(b * 512 + col)) * 4096 + row] = f2bf(val);
            } else {
                const size_t o = ((size_t)(b * 512 + col)) * 4096 + row;
                Yf[o] = val + resid[o];
            }
        }
    }
}

// ---------------------------------------------------------------------------
// Kernel 4: flash attention. Block = 16 queries, 8 waves. Q tile async-staged
// to LDS. Per 128-key tile: wave w computes S[16x16] for keys
// [kt+16w, kt+16w+16) via 16 WMMAs over c=512 (K fragments register
// double-buffered from global); parallel online softmax (16 threads/row,
// two-stage LDS reduction); P(16x128) @ V with V fragments register
// double-buffered (V pre-transposed [b,c,n] -> contiguous 128-bit loads).
// Wave w owns output channels [64w, 64w+64).
// ---------------------------------------------------------------------------
__global__ void __launch_bounds__(256)
attn_kernel(const unsigned short* __restrict__ Q,
            const unsigned short* __restrict__ Kb,
            const unsigned short* __restrict__ Vt,
            unsigned short* __restrict__ Ho) {
    __shared__ unsigned short Qs[16 * 512];   // 16 KB
    __shared__ float          Ss[16 * 128];   // 8 KB
    __shared__ unsigned short Ps[16 * 128];   // 4 KB
    __shared__ float          red[256];       // softmax partial reduce
    __shared__ float mrow[16], lrow[16], arw[16], mnewS[16];

    const int b = blockIdx.y;
    const int qbase = blockIdx.x * 16;
    const int tid = threadIdx.x, lane = tid & 31, wv = tid >> 5;
    const int hi = lane >> 4, mb = hi * 8, nl = lane & 15;
    const int srow = tid >> 4, sseg = tid & 15;     // softmax: 16 thr/row
    const float scale = 0.04419417382415922f;       // 512^-0.5

    // async-stage the contiguous 16x512 Q tile into LDS (4 x B128 per thread)
    const unsigned short* qsrc = Q + ((size_t)(b * 4096 + qbase)) * 512;
#pragma unroll
    for (int r = 0; r < 4; r++) {
        const int i = tid + r * 256;              // 1024 x 16B chunks
        async_b128(lds_addr_of(&Qs[i * 8]), qsrc + i * 8);
    }
    if (tid < 16) { mrow[tid] = -1e30f; lrow[tid] = 0.f; }

    v8f o[4];
#pragma unroll
    for (int j = 0; j < 4; j++)
#pragma unroll
        for (int i = 0; i < 8; i++) o[j][i] = 0.f;
    wait_async0();
    __syncthreads();

    for (int kt = 0; kt < 4096; kt += 128) {
        // ---- S = (Q K^T) * scale for this wave's 16 keys ----
        v8f s;
#pragma unroll
        for (int i = 0; i < 8; i++) s[i] = 0.f;
        const unsigned short* Kr =
            Kb + ((size_t)(b * 4096 + kt + wv * 16 + nl)) * 512 + (hi << 4);

        Frag kbuf[2];
        {
            const uint4* bp = (const uint4*)Kr;
            kbuf[0].q[0] = bp[0];
            kbuf[0].q[1] = bp[1];
        }
#pragma unroll
        for (int kb2 = 0; kb2 < 512; kb2 += 32) {
            const int cur = (kb2 >> 5) & 1;
            if (kb2 + 32 < 512) {           // prefetch next K fragment
                const uint4* bp = (const uint4*)(Kr + kb2 + 32);
                kbuf[cur ^ 1].q[0] = bp[0];
                kbuf[cur ^ 1].q[1] = bp[1];
            }
            Frag a;
            const uint4* ap = (const uint4*)&Qs[nl * 512 + kb2 + (hi << 4)];
            a.q[0] = ap[0];
            a.q[1] = ap[1];
            s = wmma_bf16(a, kbuf[cur], s);
        }
#pragma unroll
        for (int v = 0; v < 8; v++)
            Ss[(v + mb) * 128 + wv * 16 + nl] = s[v] * scale;
        __syncthreads();

        // ---- parallel online softmax: 16 threads per row, 8 keys each ----
        float pmax = -3.0e38f;
#pragma unroll
        for (int i = 0; i < 8; i++)
            pmax = fmaxf(pmax, Ss[srow * 128 + sseg * 8 + i]);
        red[tid] = pmax;
        __syncthreads();
        if (tid < 16) {
            float mx = mrow[tid];
            for (int i = 0; i < 16; i++) mx = fmaxf(mx, red[tid * 16 + i]);
            mnewS[tid] = mx;
        }
        __syncthreads();
        const float mnew = mnewS[srow];
        float psum = 0.f;
#pragma unroll
        for (int i = 0; i < 8; i++) {
            const float p = __expf(Ss[srow * 128 + sseg * 8 + i] - mnew);
            Ps[srow * 128 + sseg * 8 + i] = f2bf(p);
            psum += p;
        }
        red[tid] = psum;
        __syncthreads();
        if (tid < 16) {
            float lsum = 0.f;
            for (int i = 0; i < 16; i++) lsum += red[tid * 16 + i];
            const float alpha = __expf(mrow[tid] - mnewS[tid]);
            lrow[tid] = lrow[tid] * alpha + lsum;
            mrow[tid] = mnewS[tid];
            arw[tid]  = alpha;
        }
        __syncthreads();

        // ---- rescale accumulators, then O += P @ V (pipelined V frags) ----
#pragma unroll
        for (int j = 0; j < 4; j++)
#pragma unroll
            for (int v = 0; v < 8; v++) o[j][v] *= arw[v + mb];

        const unsigned short* Vg =
            Vt + ((size_t)(b * 512 + wv * 64 + nl)) * 4096 + kt + (hi << 4);
        Frag vb[2];
        {
            const uint4* vp = (const uint4*)Vg;   // (kb3=0, j=0)
            vb[0].q[0] = vp[0];
            vb[0].q[1] = vp[1];
        }
        Frag a;
#pragma unroll
        for (int t = 0; t < 16; t++) {
            const int kb3 = (t >> 2) * 32;
            const int j = t & 3;
            const int cur = t & 1;
            if (t + 1 < 16) {                    // prefetch next V fragment
                const int t2 = t + 1;
                const int kb3n = (t2 >> 2) * 32, jn = t2 & 3;
                const uint4* vp = (const uint4*)(Vg + (size_t)(jn * 16) * 4096 + kb3n);
                vb[cur ^ 1].q[0] = vp[0];
                vb[cur ^ 1].q[1] = vp[1];
            }
            if (j == 0) {
                const uint4* pp = (const uint4*)&Ps[nl * 128 + kb3 + (hi << 4)];
                a.q[0] = pp[0];
                a.q[1] = pp[1];
            }
            o[j] = wmma_bf16(a, vb[cur], o[j]);
        }
        __syncthreads();
    }

    // ---- normalize and store Ho[b][n][c] bf16 ----
#pragma unroll
    for (int j = 0; j < 4; j++) {
        const int col = wv * 64 + j * 16 + nl;
#pragma unroll
        for (int v = 0; v < 8; v++) {
            const int r = v + mb;
            const float val = o[j][v] / lrow[r];
            Ho[((size_t)(b * 4096 + qbase + r)) * 512 + col] = f2bf(val);
        }
    }
}

// ---------------------------------------------------------------------------
// Host-side orchestration
// ---------------------------------------------------------------------------
extern "C" void kernel_launch(void* const* d_in, const int* in_sizes, int n_in,
                              void* d_out, int out_size, void* d_ws, size_t ws_size,
                              hipStream_t stream) {
    const float* X    = (const float*)d_in[0];
    const float* gn_w = (const float*)d_in[1];
    const float* gn_b = (const float*)d_in[2];
    const float* wq   = (const float*)d_in[3];
    const float* bq   = (const float*)d_in[4];
    const float* wk   = (const float*)d_in[5];
    const float* bk   = (const float*)d_in[6];
    const float* wvp  = (const float*)d_in[7];
    const float* bv   = (const float*)d_in[8];
    const float* wp   = (const float*)d_in[9];
    const float* bp   = (const float*)d_in[10];
    float* out = (float*)d_out;

    const size_t BNC = (size_t)4 * 4096 * 512;   // 8,388,608 elements
    const size_t WSZ = (size_t)512 * 512;        // 262,144 elements

    unsigned short* Ht  = (unsigned short*)d_ws;
    unsigned short* Qb  = Ht  + BNC;
    unsigned short* Kbf = Qb  + BNC;
    unsigned short* Vt  = Kbf + BNC;             // stored [b,c,n]
    unsigned short* Hob = Vt  + BNC;
    unsigned short* Wq  = Hob + BNC;
    unsigned short* Wk  = Wq  + WSZ;
    unsigned short* Wv  = Wk  + WSZ;
    unsigned short* Wp  = Wv  + WSZ;

    // 1) weights -> bf16
    cvt_bf16_kernel<<<512, 256, 0, stream>>>(wq,  Wq, (int)WSZ);
    cvt_bf16_kernel<<<512, 256, 0, stream>>>(wk,  Wk, (int)WSZ);
    cvt_bf16_kernel<<<512, 256, 0, stream>>>(wvp, Wv, (int)WSZ);
    cvt_bf16_kernel<<<512, 256, 0, stream>>>(wp,  Wp, (int)WSZ);

    // 2) GroupNorm -> Ht[b,n,c] bf16
    groupnorm_kernel<<<128, 256, 0, stream>>>(X, gn_w, gn_b, Ht);

    // 3) Q / K / V projections
    dim3 ggrid(64, 4, 4);
    gemm_kernel<<<ggrid, 256, 0, stream>>>(Ht, Wq, bq, nullptr, Qb,  nullptr, 0);
    gemm_kernel<<<ggrid, 256, 0, stream>>>(Ht, Wk, bk, nullptr, Kbf, nullptr, 0);
    gemm_kernel<<<ggrid, 256, 0, stream>>>(Ht, Wv, bv, nullptr, Vt,  nullptr, 2);

    // 4) flash attention -> Hob[b,n,c] bf16
    attn_kernel<<<dim3(256, 4), 256, 0, stream>>>(Qb, Kbf, Vt, Hob);

    // 5) proj_out + residual -> d_out fp32 [b,c,h,w]
    gemm_kernel<<<ggrid, 256, 0, stream>>>(Hob, Wp, bp, X, nullptr, out, 1);
}
